// SimpleRNN_37984690766359
// MI455X (gfx1250) — matrix operand order repaired and checked by reading, not compiled, not run on hardware
//
#include <hip/hip_runtime.h>
#include <cstdint>

#define B_   64
#define S_   2048
#define IN_  256
#define H_   128
#define OUT_ 3

typedef __attribute__((ext_vector_type(16))) __bf16       v16bf;
typedef __attribute__((ext_vector_type(2)))  __bf16       v2bf;
typedef __attribute__((ext_vector_type(8)))  float        v8f;
typedef __attribute__((ext_vector_type(8)))  unsigned int v8u;

// f32x2 -> packed bf16x2. Prefer hardware cvt; native __bf16 casts let ISel
// pick v_cvt_pk_bf16_f32 if present, else clang emits the RTNE sequence.
#if __has_builtin(__builtin_amdgcn_cvt_pk_bf16_f32)
static __device__ __forceinline__ unsigned pk2(float a, float b) {
  return __builtin_bit_cast(unsigned, __builtin_amdgcn_cvt_pk_bf16_f32(a, b));
}
#else
static __device__ __forceinline__ unsigned pk2(float a, float b) {
  v2bf p;
  p[0] = (__bf16)a;
  p[1] = (__bf16)b;
  return __builtin_bit_cast(unsigned, p);
}
#endif

// Prefer hardware V_TANH_F32 (gfx1250 trans op) over OCML expansion.
#if __has_builtin(__builtin_amdgcn_tanhf)
#define TANH_(x) __builtin_amdgcn_tanhf(x)
#elif __has_builtin(__builtin_amdgcn_tanh_f32)
#define TANH_(x) __builtin_amdgcn_tanh_f32(x)
#else
#define TANH_(x) tanhf(x)
#endif

static __device__ __forceinline__ v16bf as_bf(v8u u) {
  return __builtin_bit_cast(v16bf, u);
}

// ---------------------------------------------------------------------------
// Phase 1: xproj[S,B,H] = x[B,S,IN] @ W_xh^T + b_xh   (written transposed)
// 8 waves/WG; each wave: one 16-row M-tile x all 8 N-tiles (full H=128).
// W_xh staged once in LDS as bf16, B-operand-friendly, padded rows.
// ---------------------------------------------------------------------------
#define WSTR 132  // u32 per LDS row of W (128 data + 4 pad -> bank-spread)

__global__ __launch_bounds__(256) void xproj_kernel(
    const float* __restrict__ x, const float* __restrict__ W_xh,
    const float* __restrict__ b_xh, float* __restrict__ xproj) {
  __shared__ unsigned ldsW[H_ * WSTR];
  const int lane = threadIdx.x & 31;
  const int wave = threadIdx.x >> 5;
  const int nl   = lane & 15;
  const int hi   = lane >> 4;

  // stage W_xh -> LDS as packed bf16 pairs
  const float2* w2 = (const float2*)W_xh;
  for (int i = threadIdx.x; i < H_ * (IN_ / 2); i += 256) {
    int n = i >> 7;
    int p = i & 127;
    float2 v = w2[n * (IN_ / 2) + p];
    ldsW[n * WSTR + p] = pk2(v.x, v.y);
  }
  __syncthreads();

  float biasj[8];
#pragma unroll
  for (int j = 0; j < 8; ++j) biasj[j] = b_xh[16 * j + nl];

  const int wgid = blockIdx.x * 8 + wave;  // 0..4095 ; 8192 m-tiles total
#pragma unroll
  for (int rep = 0; rep < 2; ++rep) {
    const int mt = wgid + rep * 4096;
    const long m = (long)mt * 16 + nl;
    const float* xrow = x + m * IN_;

    v8f C[8] = {};
    for (int c = 0; c < 8; ++c) {  // K chunks of 32
      const float4* ap = (const float4*)(xrow + 32 * c + 8 * hi);
      float4 a0 = ap[0], a1 = ap[1], a2 = ap[4], a3 = ap[5];
      v8u au = {pk2(a0.x, a0.y), pk2(a0.z, a0.w), pk2(a1.x, a1.y), pk2(a1.z, a1.w),
                pk2(a2.x, a2.y), pk2(a2.z, a2.w), pk2(a3.x, a3.y), pk2(a3.z, a3.w)};
      v16bf A = as_bf(au);
#pragma unroll
      for (int j = 0; j < 8; ++j) {  // all 8 N-tiles reuse A
        const uint4* bp =
            (const uint4*)&ldsW[(16 * j + nl) * WSTR + 16 * c + 4 * hi];
        uint4 b0 = bp[0], b1 = bp[2];
        v8u bu = {b0.x, b0.y, b0.z, b0.w, b1.x, b1.y, b1.z, b1.w};
        C[j] = __builtin_amdgcn_wmma_f32_16x16x32_bf16(
            false, A, false, as_bf(bu), (short)0, C[j], false, false);
      }
    }
    const int b  = mt >> 7;          // 128 m-tiles per batch row
    const int t0 = (mt & 127) * 16;
#pragma unroll
    for (int j = 0; j < 8; ++j) {
#pragma unroll
      for (int r = 0; r < 8; ++r) {
        int t = t0 + r + 8 * hi;     // C layout: row = r + 8*half
        xproj[((long)t * B_ + b) * H_ + 16 * j + nl] = C[j][r] + biasj[j];
      }
    }
  }
}

// ---------------------------------------------------------------------------
// Phase 2: sequential scan, 4 WGs (16 batch rows) x 8 waves (16 cols).
// W_hh^T resident in registers (bf16 B operands); h ping-pong in LDS (bf16);
// one barrier per step; xp_t stream software-pipelined one step ahead.
// ---------------------------------------------------------------------------
#define HSTR 68  // u32 per LDS row of h (64 data + 4 pad)

__global__ __launch_bounds__(256) void rnn_scan_kernel(
    const float* __restrict__ xproj, const float* __restrict__ W_hh,
    const float* __restrict__ b_hh, const float* __restrict__ b_h,
    float* __restrict__ hfinal) {
  __shared__ unsigned ldsH[2][16 * HSTR];
  const int lane = threadIdx.x & 31;
  const int wave = threadIdx.x >> 5;
  const int nl   = lane & 15;
  const int hi   = lane >> 4;
  const int b0   = blockIdx.x * 16;
  const int col  = wave * 16 + nl;

  // B = W_hh^T, bf16, resident in registers (4 chunks x 8 VGPRs)
  v16bf Bv[4];
#pragma unroll
  for (int c = 0; c < 4; ++c) {
    const float* wr = W_hh + (long)col * H_ + 32 * c + 16 * hi;
    v8u bu;
#pragma unroll
    for (int v = 0; v < 8; ++v) bu[v] = pk2(wr[2 * v], wr[2 * v + 1]);
    Bv[c] = as_bf(bu);
  }
  const float bias = b_hh[col] + b_h[col];

  for (int i = threadIdx.x; i < 16 * HSTR; i += 256) ldsH[0][i] = 0;  // h0 = 0
  __syncthreads();

  // xp base for this lane: rows b0+8*hi+r, column col; +t*B_*H_ per step
  const float* xb = xproj + (long)(b0 + 8 * hi) * H_ + col;
  float xn[8];
#pragma unroll
  for (int r = 0; r < 8; ++r) xn[r] = xb[r * H_];  // prefetch t = 0

  v8f C = {};
  int pb = 0;
  for (int t = 0; t < S_; ++t) {
    float xv[8];
#pragma unroll
    for (int r = 0; r < 8; ++r) xv[r] = xn[r];
    // prefetch next step's xp (hides L2 latency behind this step's compute)
    const long toff = (long)((t + 1 < S_) ? t + 1 : t) * (B_ * H_);
#pragma unroll
    for (int r = 0; r < 8; ++r) xn[r] = xb[toff + r * H_];

    // A = h (bf16) from LDS, contiguous k order (k = 16*half + s), same as Bv
    v16bf A[4];
#pragma unroll
    for (int c = 0; c < 4; ++c) {
      const uint4* p = (const uint4*)&ldsH[pb][nl * HSTR + 16 * c + 8 * hi];
      uint4 lo = p[0], h2 = p[1];
      v8u au = {lo.x, lo.y, lo.z, lo.w, h2.x, h2.y, h2.z, h2.w};
      A[c] = as_bf(au);
    }

#pragma unroll
    for (int r = 0; r < 8; ++r) C[r] = xv[r] + bias;
#pragma unroll
    for (int c = 0; c < 4; ++c)
      C = __builtin_amdgcn_wmma_f32_16x16x32_bf16(false, A[c], false, Bv[c],
                                                  (short)0, C, false, false);
#pragma unroll
    for (int r = 0; r < 8; ++r) C[r] = TANH_(C[r]);

    // write h_{t+1} (bf16) into the other buffer; single barrier per step
    __bf16* hp = (__bf16*)ldsH[pb ^ 1];
#pragma unroll
    for (int r = 0; r < 8; ++r)
      hp[(r + 8 * hi) * (2 * HSTR) + col] = (__bf16)C[r];
    __syncthreads();
    pb ^= 1;
  }

#pragma unroll
  for (int r = 0; r < 8; ++r)
    hfinal[(long)(b0 + r + 8 * hi) * H_ + col] = C[r];
}

// ---------------------------------------------------------------------------
// Phase 3: out[64,3] = hfinal @ W_fc^T + b_fc  (tiny)
// ---------------------------------------------------------------------------
__global__ void head_kernel(const float* __restrict__ hfinal,
                            const float* __restrict__ W_fc,
                            const float* __restrict__ b_fc,
                            float* __restrict__ out) {
  int tid = threadIdx.x;
  if (tid >= B_ * OUT_) return;
  int b = tid / OUT_, o = tid % OUT_;
  float s = b_fc[o];
  for (int k = 0; k < H_; ++k) s += hfinal[b * H_ + k] * W_fc[o * H_ + k];
  out[tid] = s;
}

extern "C" void kernel_launch(void* const* d_in, const int* in_sizes, int n_in,
                              void* d_out, int out_size, void* d_ws,
                              size_t ws_size, hipStream_t stream) {
  const float* x    = (const float*)d_in[0];
  const float* W_xh = (const float*)d_in[1];
  const float* b_xh = (const float*)d_in[2];
  const float* W_hh = (const float*)d_in[3];
  const float* b_hh = (const float*)d_in[4];
  const float* b_h  = (const float*)d_in[5];
  const float* W_fc = (const float*)d_in[6];
  const float* b_fc = (const float*)d_in[7];
  float* out = (float*)d_out;

  float* xproj  = (float*)d_ws;                   // [S,B,H] f32 (64 MB)
  float* hfinal = xproj + (size_t)S_ * B_ * H_;   // [B,H] f32

  xproj_kernel<<<512, 256, 0, stream>>>(x, W_xh, b_xh, xproj);
  rnn_scan_kernel<<<4, 256, 0, stream>>>(xproj, W_hh, b_hh, b_h, hfinal);
  head_kernel<<<1, 256, 0, stream>>>(hfinal, W_fc, b_fc, out);
}
